// Encoders_5454608466563
// MI455X (gfx1250) — matrix operand order repaired and checked by reading, not compiled, not run on hardware
//
#include <hip/hip_runtime.h>
#include <hip/hip_bf16.h>

typedef __bf16 bf16;
typedef __attribute__((ext_vector_type(16))) __bf16 v16bf;
typedef __attribute__((ext_vector_type(8)))  __bf16 bf16x8;
typedef __attribute__((ext_vector_type(8)))  float  v8f;

// ---------------------------------------------------------------------------
// gfx1250 async global->LDS 16B copy (VFLAT GLOBAL_LOAD_ASYNC_TO_LDS_B128,
// GV addressing, tracked by ASYNCcnt). LDS operand is the low 32 bits of the
// generic pointer.
// ---------------------------------------------------------------------------
__device__ __forceinline__ void async_copy_b128(void* ldsDst, const void* gSrc)
{
    asm volatile("global_load_async_to_lds_b128 %0, %1, off"
                 :: "v"((unsigned)(unsigned long long)ldsDst), "v"(gSrc)
                 : "memory");
}

template<int N>
__device__ __forceinline__ void wait_asynccnt()
{
    if constexpr      (N == 0) asm volatile("s_wait_asynccnt 0" ::: "memory");
    else if constexpr (N == 1) asm volatile("s_wait_asynccnt 1" ::: "memory");
    else if constexpr (N == 2) asm volatile("s_wait_asynccnt 2" ::: "memory");
    else if constexpr (N == 3) asm volatile("s_wait_asynccnt 3" ::: "memory");
    else if constexpr (N == 4) asm volatile("s_wait_asynccnt 4" ::: "memory");
    else                       asm volatile("s_wait_asynccnt 6" ::: "memory");
}

// ---------------------------------------------------------------------------
// Tiled bf16 WMMA GEMM with double-buffered async-LDS staging.
//   A  : bf16 [M,K] row-major (lda), per-z base = (z/Hh)*aBatch + (z%Hh)*aHead
//   Bt : bf16 [N,K] row-major (ldb) (pre-transposed weights / Q rows / V^T)
//   EPI: 0 f32+bias | 1 bf16+bias+ReLU | 2 f32 logits (+maskT*NEG)*0.125
//        3 bf16 plain | 4 bf16+bias | 5 bf16+bias, V^T per-head scatter
// Requires M%BM==0, N%BN==0, K%32==0 (all uses satisfy this).
// ---------------------------------------------------------------------------
template<int BM, int BN, int WM_WAVES, int WN_WAVES, int EPI>
__global__ __launch_bounds__(32 * WM_WAVES * WN_WAVES)
void wmma_gemm(const bf16* __restrict__ Abase, long aBatch, long aHead, int lda,
               const bf16* __restrict__ Bbase, long bBatch, long bHead, int ldb,
               void* __restrict__ Cbase, long cBatch, long cHead, int ldc,
               const float* __restrict__ bias,
               const float* __restrict__ maskBase, long maskBatch,
               int M, int N, int K, int Hh, int Sdim)
{
    constexpr int BK  = 32;
    constexpr int NT  = 32 * WM_WAVES * WN_WAVES;
    constexpr int WM  = BM / WM_WAVES;
    constexpr int WN  = BN / WN_WAVES;
    constexpr int TM  = WM / 16;
    constexpr int TN  = WN / 16;
    constexpr int LDK = BK + 8;               // 80B rows: 16B-aligned chunks, fewer conflicts
    constexpr int IA  = (BM * BK / 8) / NT;   // 16B chunks of A per thread per tile
    constexpr int IB  = (BN * BK / 8) / NT;

    __shared__ __attribute__((aligned(16))) bf16 sA[2][BM][LDK];
    __shared__ __attribute__((aligned(16))) bf16 sB[2][BN][LDK];

    const int z    = blockIdx.z;
    const int bIdx = z / Hh;
    const int hIdx = z % Hh;
    const bf16* A  = Abase + (long)bIdx * aBatch + (long)hIdx * aHead;
    const bf16* Bt = Bbase + (long)bIdx * bBatch + (long)hIdx * bHead;
    void* C        = (void*)((char*)Cbase +
                     sizeof(float) * 0);      // element size applied per-EPI below
    const long cz  = (long)bIdx * cBatch + (long)hIdx * cHead;

    const int tid      = threadIdx.x;
    const int wave     = tid >> 5;
    const int lane     = tid & 31;
    const int wm       = wave / WN_WAVES;
    const int wn       = wave % WN_WAVES;
    const int laneHalf = lane >> 4;
    const int lane16   = lane & 15;

    const int m0 = blockIdx.y * BM;
    const int n0 = blockIdx.x * BN;

    v8f acc[TM][TN] = {};

    auto load_tiles = [&](int buf, int k0) {
        #pragma unroll
        for (int it = 0; it < IA; ++it) {
            int c = tid + it * NT;
            int r = c >> 2, cc = (c & 3) * 8;          // BK/8 == 4 chunks per row
            async_copy_b128(&sA[buf][r][cc], &A[(long)(m0 + r) * lda + k0 + cc]);
        }
        #pragma unroll
        for (int it = 0; it < IB; ++it) {
            int c = tid + it * NT;
            int r = c >> 2, cc = (c & 3) * 8;
            async_copy_b128(&sB[buf][r][cc], &Bt[(long)(n0 + r) * ldb + k0 + cc]);
        }
    };

    const int nsteps = K / BK;
    load_tiles(0, 0);

    for (int step = 0; step < nsteps; ++step) {
        const int cur = step & 1;
        if (step + 1 < nsteps) {
            load_tiles(cur ^ 1, (step + 1) * BK);      // prefetch next tile
            wait_asynccnt<IA + IB>();                  // current tile landed
        } else {
            wait_asynccnt<0>();
        }
        __syncthreads();

        v16bf afrag[TM], bfrag[TN];
        #pragma unroll
        for (int t = 0; t < TM; ++t) {
            const bf16* p = &sA[cur][wm * WM + t * 16 + lane16][laneHalf * 8];
            bf16x8 lo = *(const bf16x8*)p;             // K = half*8 .. +7
            bf16x8 hi = *(const bf16x8*)(p + 16);      // K = 16+half*8 .. +7
            afrag[t] = __builtin_shufflevector(lo, hi, 0,1,2,3,4,5,6,7,
                                                       8,9,10,11,12,13,14,15);
        }
        #pragma unroll
        for (int u = 0; u < TN; ++u) {
            const bf16* p = &sB[cur][wn * WN + u * 16 + lane16][laneHalf * 8];
            bf16x8 lo = *(const bf16x8*)p;             // K = half*8 .. +7
            bf16x8 hi = *(const bf16x8*)(p + 16);      // K = 16+half*8 .. +7
            bfrag[u] = __builtin_shufflevector(lo, hi, 0,1,2,3,4,5,6,7,
                                                       8,9,10,11,12,13,14,15);
        }
        #pragma unroll
        for (int t = 0; t < TM; ++t)
            #pragma unroll
            for (int u = 0; u < TN; ++u)
                acc[t][u] = __builtin_amdgcn_wmma_f32_16x16x32_bf16(
                    false, afrag[t], false, bfrag[u], (short)0, acc[t][u],
                    false, false);
        __syncthreads();
    }

    // ---- epilogue ----
    const float* mz = (EPI == 2) ? (maskBase + (long)bIdx * maskBatch) : nullptr;
    #pragma unroll
    for (int t = 0; t < TM; ++t) {
        #pragma unroll
        for (int u = 0; u < TN; ++u) {
            #pragma unroll
            for (int r = 0; r < 8; ++r) {
                const int row = m0 + wm * WM + t * 16 + laneHalf * 8 + r;
                const int col = n0 + wn * WN + u * 16 + lane16;
                const float val = acc[t][u][r];
                if constexpr (EPI == 0) {
                    ((float*)C)[cz + (long)row * ldc + col] = val + bias[col];
                } else if constexpr (EPI == 1) {
                    ((bf16*)C)[cz + (long)row * ldc + col] =
                        (bf16)fmaxf(val + bias[col], 0.f);
                } else if constexpr (EPI == 2) {
                    ((float*)C)[cz + (long)row * ldc + col] =
                        (val + mz[(long)col * M + row] * (-1.0e9f)) * 0.125f;
                } else if constexpr (EPI == 3) {
                    ((bf16*)C)[cz + (long)row * ldc + col] = (bf16)val;
                } else if constexpr (EPI == 4) {
                    ((bf16*)C)[cz + (long)row * ldc + col] = (bf16)(val + bias[col]);
                } else {                                  // EPI 5: V^T scatter
                    const int b = row / Sdim, i = row % Sdim;
                    const int h = col >> 6,  d = col & 63;
                    ((bf16*)C)[((((long)b * (N >> 6) + h) << 6) + d) * Sdim + i] =
                        (bf16)(val + bias[col]);
                }
            }
        }
    }
}

// ---------------------------------------------------------------------------
__global__ __launch_bounds__(256)
void cvt_bf16(const float* __restrict__ in, bf16* __restrict__ out, long n)
{
    long i = (long)blockIdx.x * 256 + threadIdx.x;
    if (i < n) out[i] = (bf16)in[i];
}

// Wt[n][k] = (bf16)W[k][n]
__global__ __launch_bounds__(256)
void transpose_cvt(const float* __restrict__ W, bf16* __restrict__ Wt, int K, int N)
{
    __shared__ float t[32][33];
    const int k0 = blockIdx.y * 32, n0 = blockIdx.x * 32;
    const int tx = threadIdx.x & 31, ty = threadIdx.x >> 5;   // 32 x 8
    for (int i = ty; i < 32; i += 8) t[i][tx] = W[(long)(k0 + i) * N + n0 + tx];
    __syncthreads();
    for (int i = ty; i < 32; i += 8)
        Wt[(long)(n0 + i) * K + k0 + tx] = (bf16)t[tx][i];
}

// ---------------------------------------------------------------------------
__global__ __launch_bounds__(256)
void softmax_stats(const float* __restrict__ logits, float* __restrict__ stats, int n)
{
    const int z = blockIdx.x;
    const float* p = logits + (long)z * n;
    __shared__ float red[256];
    const int tid = threadIdx.x;

    float mx = -3.4e38f;
    for (int i = tid; i < n; i += 256) mx = fmaxf(mx, p[i]);
    red[tid] = mx; __syncthreads();
    for (int s = 128; s > 0; s >>= 1) {
        if (tid < s) red[tid] = fmaxf(red[tid], red[tid + s]);
        __syncthreads();
    }
    mx = red[0]; __syncthreads();

    float sum = 0.f;
    for (int i = tid; i < n; i += 256) sum += __expf(p[i] - mx);
    red[tid] = sum; __syncthreads();
    for (int s = 128; s > 0; s >>= 1) {
        if (tid < s) red[tid] += red[tid + s];
        __syncthreads();
    }
    if (tid == 0) { stats[z * 2] = mx; stats[z * 2 + 1] = red[0]; }
}

__global__ __launch_bounds__(256)
void softmax_norm(const float* __restrict__ logits, const float* __restrict__ stats,
                  bf16* __restrict__ attn, long n)
{
    const int z = blockIdx.z;
    const float* p = logits + (long)z * n;
    bf16* o = attn + (long)z * n;
    const float mx  = stats[z * 2];
    const float inv = 1.f / stats[z * 2 + 1];
    long i = (long)blockIdx.x * 256 + threadIdx.x;
    if (i < n) o[i] = (bf16)(__expf(p[i] - mx) * inv);
}

// ---------------------------------------------------------------------------
// out = LN(res + delta)*gamma + beta  (f32, plus optional bf16 twin). D == 512.
// ---------------------------------------------------------------------------
__global__ __launch_bounds__(256)
void add_layernorm(const float* __restrict__ res, const float* __restrict__ delta,
                   const float* __restrict__ gamma, const float* __restrict__ beta,
                   float* __restrict__ out, bf16* __restrict__ outbf, int D)
{
    const long row = blockIdx.x;
    const float* r = res   + row * D;
    const float* d = delta + row * D;
    __shared__ float red[256], red2[256];
    const int tid = threadIdx.x;

    float s = 0.f, s2 = 0.f;
    float vals[2];
    for (int i = tid, c = 0; i < D; i += 256, ++c) {
        float x = r[i] + d[i];
        vals[c] = x; s += x; s2 += x * x;
    }
    red[tid] = s; red2[tid] = s2; __syncthreads();
    for (int st = 128; st > 0; st >>= 1) {
        if (tid < st) { red[tid] += red[tid + st]; red2[tid] += red2[tid + st]; }
        __syncthreads();
    }
    const float mean = red[0] / D;
    const float var  = red2[0] / D - mean * mean;
    const float rstd = rsqrtf(var + 1e-9f);
    for (int i = tid, c = 0; i < D; i += 256, ++c) {
        float o = gamma[i] * (vals[c] - mean) * rstd + beta[i];
        out[row * D + i] = o;
        if (outbf) outbf[row * D + i] = (bf16)o;
    }
}

// ---------------------------------------------------------------------------
extern "C" void kernel_launch(void* const* d_in, const int* in_sizes, int n_in,
                              void* d_out, int out_size, void* d_ws, size_t ws_size,
                              hipStream_t stream)
{
    constexpr int  Bb = 4, S = 1024, D = 512, Hh = 8, DFF = 2048;
    constexpr long TOK = (long)Bb * S * D;           // 2,097,152

    const float* x    = (const float*)d_in[0];
    const float* mask = (const float*)d_in[1];
    const float* Wq = (const float*)d_in[2];  const float* bq  = (const float*)d_in[3];
    const float* Wk = (const float*)d_in[4];  const float* bk  = (const float*)d_in[5];
    const float* Wv = (const float*)d_in[6];  const float* bv  = (const float*)d_in[7];
    const float* Wo = (const float*)d_in[8];  const float* bo  = (const float*)d_in[9];
    const float* W1 = (const float*)d_in[10]; const float* b1  = (const float*)d_in[11];
    const float* W2 = (const float*)d_in[12]; const float* b2  = (const float*)d_in[13];
    const float* g1 = (const float*)d_in[14]; const float* be1 = (const float*)d_in[15];
    const float* g2 = (const float*)d_in[16]; const float* be2 = (const float*)d_in[17];

    float* p = (float*)d_ws;
    auto alloc_f = [&](long n) { float* r = p; p += n; return r; };
    bf16*  xbf    = (bf16*)alloc_f(TOK / 2);
    bf16*  qbf    = (bf16*)alloc_f(TOK / 2);
    bf16*  kbf    = (bf16*)alloc_f(TOK / 2);
    bf16*  vT     = (bf16*)alloc_f(TOK / 2);             // [B][H][64][S]
    bf16*  ctxbf  = (bf16*)alloc_f(TOK / 2);
    bf16*  h1bf   = (bf16*)alloc_f(TOK / 2);
    bf16*  hbufbf = (bf16*)alloc_f(TOK / 2);
    float* h1     = alloc_f(TOK);
    float* hbuf   = alloc_f(TOK);
    float* proj   = alloc_f(TOK);                        // Wo out / FFN2 out
    bf16*  ffn1bf = (bf16*)alloc_f((long)Bb * S * DFF / 2);
    float* logits = alloc_f((long)Bb * Hh * S * S);      // 16*TOK
    bf16*  attnbf = (bf16*)alloc_f((long)Bb * Hh * S * S / 2);
    bf16*  Wqt = (bf16*)alloc_f((long)D * D / 2);
    bf16*  Wkt = (bf16*)alloc_f((long)D * D / 2);
    bf16*  Wvt = (bf16*)alloc_f((long)D * D / 2);
    bf16*  Wot = (bf16*)alloc_f((long)D * D / 2);
    bf16*  W1t = (bf16*)alloc_f((long)D * DFF / 2);      // [DFF][D]
    bf16*  W2t = (bf16*)alloc_f((long)D * DFF / 2);      // [D][DFF]
    float* stats = alloc_f(64);

    const dim3 blk(256);

    // ---- one-time conversions (deterministic each call) ----
    cvt_bf16<<<dim3(TOK / 256), blk, 0, stream>>>(x, xbf, TOK);
    transpose_cvt<<<dim3(D / 32,   D / 32),   blk, 0, stream>>>(Wq, Wqt, D, D);
    transpose_cvt<<<dim3(D / 32,   D / 32),   blk, 0, stream>>>(Wk, Wkt, D, D);
    transpose_cvt<<<dim3(D / 32,   D / 32),   blk, 0, stream>>>(Wv, Wvt, D, D);
    transpose_cvt<<<dim3(D / 32,   D / 32),   blk, 0, stream>>>(Wo, Wot, D, D);
    transpose_cvt<<<dim3(DFF / 32, D / 32),   blk, 0, stream>>>(W1, W1t, D, DFF);
    transpose_cvt<<<dim3(D / 32,   DFF / 32), blk, 0, stream>>>(W2, W2t, DFF, D);

    const float* hin   = x;
    const bf16*  hinbf = xbf;

    for (int layer = 0; layer < 2; ++layer) {
        // ---- Q, K projections (bf16 out) ; V projection scatters V^T ----
        wmma_gemm<128,128,4,2,4><<<dim3(D/128, (Bb*S)/128, 1), blk, 0, stream>>>(
            hinbf, 0,0, D,  Wqt, 0,0, D,  qbf, 0,0, D,  bq, nullptr, 0,
            Bb*S, D, D, 1, S);
        wmma_gemm<128,128,4,2,4><<<dim3(D/128, (Bb*S)/128, 1), blk, 0, stream>>>(
            hinbf, 0,0, D,  Wkt, 0,0, D,  kbf, 0,0, D,  bk, nullptr, 0,
            Bb*S, D, D, 1, S);
        wmma_gemm<128,128,4,2,5><<<dim3(D/128, (Bb*S)/128, 1), blk, 0, stream>>>(
            hinbf, 0,0, D,  Wvt, 0,0, D,  vT, 0,0, D,  bv, nullptr, 0,
            Bb*S, D, D, 1, S);

        // ---- transposed logits: Lt[j,i] = k_j . q_i  (A=K rows, Bt=Q rows) ----
        wmma_gemm<128,128,4,2,2><<<dim3(S/128, S/128, Bb*Hh), blk, 0, stream>>>(
            kbf, (long)S*D, 64, D,
            qbf, (long)S*D, 64, D,
            logits, (long)Hh*S*S, (long)S*S, S,
            nullptr, mask, (long)S*S, S, S, 64, Hh, S);

        // ---- joint softmax over S*S per (b,h) ----
        softmax_stats<<<Bb*Hh, blk, 0, stream>>>(logits, stats, S*S);
        softmax_norm<<<dim3((S*S)/256, 1, Bb*Hh), blk, 0, stream>>>(
            logits, stats, attnbf, (long)S*S);

        // ---- ctx[j,d] = sum_i attnT[j,i] * v[i,d]  (Bt = V^T rows) ----
        wmma_gemm<128,64,8,1,3><<<dim3(1, S/128, Bb*Hh), blk, 0, stream>>>(
            attnbf, (long)Hh*S*S, (long)S*S, S,
            vT,     (long)Hh*64*S, (long)64*S, S,
            ctxbf,  (long)S*D,    64,        D,
            nullptr, nullptr, 0, S, 64, S, Hh, S);

        // ---- output projection -> f32 ----
        wmma_gemm<128,128,4,2,0><<<dim3(D/128, (Bb*S)/128, 1), blk, 0, stream>>>(
            ctxbf, 0,0, D,  Wot, 0,0, D,  proj, 0,0, D,  bo, nullptr, 0,
            Bb*S, D, D, 1, S);

        // ---- h1 = LN(hin + attn_out) ----
        add_layernorm<<<Bb*S, blk, 0, stream>>>(hin, proj, g1, be1, h1, h1bf, D);

        // ---- FFN ----
        wmma_gemm<128,128,4,2,1><<<dim3(DFF/128, (Bb*S)/128, 1), blk, 0, stream>>>(
            h1bf, 0,0, D,  W1t, 0,0, D,  ffn1bf, 0,0, DFF,  b1, nullptr, 0,
            Bb*S, DFF, D, 1, S);
        wmma_gemm<128,128,4,2,0><<<dim3(D/128, (Bb*S)/128, 1), blk, 0, stream>>>(
            ffn1bf, 0,0, DFF,  W2t, 0,0, DFF,  proj, 0,0, D,  b2, nullptr, 0,
            Bb*S, D, DFF, 1, S);

        // ---- h = LN(h1 + ffn) ----
        float* outp  = (layer == 1) ? (float*)d_out : hbuf;
        bf16*  outbf = (layer == 1) ? (bf16*)nullptr : hbufbf;
        add_layernorm<<<Bb*S, blk, 0, stream>>>(h1, proj, g2, be2, outp, outbf, D);
        hin = hbuf; hinbf = hbufbf;
    }
}
